// Eq1dConv_22101901705657
// MI455X (gfx1250) — compile-verified, matrix operand order checked by reading, not run on hardware
//
#include <hip/hip_runtime.h>
#include <hip/hip_bf16.h>

// ---------------------------------------------------------------------------
// Fused: 1x3 conv (CIN=64 -> COUT=64, pad 1 in W) + bias
//        + polyphase [up2 / 1x5 FIR (gain4)] -> leakyReLU(0.2) -> [1x4 FIR / down2]
// H axis of the up/down stages is identity; per-(b,h)-row 1D pipeline along W.
// Conv as bf16 WMMA GEMM (M=cout, N=positions, K=cin*3=192), fp32 accumulate.
// x tile transposed [pos][cin] in LDS  -> B fragments: 12x ds_load_b128 batched
//   per pos-tile (distinct regs -> staged dscnt waits, latency paid once).
// Weights pre-swizzled to A-fragment layout, hoisted to VGPRs per wave.
// y buffer transposed [pos][cout]      -> epilogue: 2x ds_store_b128.
// ---------------------------------------------------------------------------

#define B_    8
#define CIN_  64
#define COUT_ 64
#define H_    64
#define W_    512
#define TILE_ 128          // output positions per workgroup
#define NPT_  10           // pos-tiles computed (160 incl. halo + split slack)
#define XQ_   162          // x positions staged (NPT*16 + 2 conv halo)
#define CSTR_ 72           // xT row stride, bf16 elems (144B, 16B-aligned)
#define YROWS_ 146         // y rows kept (positions p0 .. p0+145)
#define YSTR_ 68           // y row stride, f32 elems (272B, 16B-aligned)
#define SLOPE_ 0.2f

#define SX_BYTES (XQ_ * CSTR_ * 2)        // 23328
#define SY_BYTES (YROWS_ * YSTR_ * 4)     // 39712 (A-swizzle 24576 aliased inside)

typedef __attribute__((ext_vector_type(16))) __bf16 v16bf;
typedef __attribute__((ext_vector_type(8)))  __bf16 v8bf;
typedef __attribute__((ext_vector_type(2)))  __bf16 v2bf;
typedef __attribute__((ext_vector_type(8)))  float  v8f;
typedef __attribute__((ext_vector_type(4)))  float  v4f;

// LDS element offset of B chunk kk relative to &sxT[(pt*16+n)*CSTR_ + 8*half]
#define BOFF(kk) (((kk) >> 1) * CSTR_ + ((kk) & 1) * 32)

__global__ __launch_bounds__(256)
void fused_conv_upfirdn_kernel(const float* __restrict__ x,
                               const float* __restrict__ wgt,   // [COUT][CIN][1][3]
                               const float* __restrict__ bias,  // [COUT]
                               const float* __restrict__ upf,   // [5]
                               const float* __restrict__ dnf,   // [4]
                               float* __restrict__ out)
{
    __shared__ __attribute__((aligned(32))) char smem[SX_BYTES + SY_BYTES];
    __bf16* sxT = (__bf16*)smem;                 // [XQ_][CSTR_] : x transposed
    __bf16* sA  = (__bf16*)(smem + SX_BYTES);    // A swizzle (dead after hoist)
    float*  sy  = (float*)(smem + SX_BYTES);     // [pos][cout] y rows (aliases sA)

    const int tile = blockIdx.x;                 // 0..3
    const int h    = blockIdx.y;
    const int b    = blockIdx.z;
    const int tid  = threadIdx.x;
    const int p0   = tile * TILE_ - 8;           // first y position held in sy

    // ---- stage 0a: global x -> LDS bf16, transposed, zero-padded -----------
    // two cin per thread -> v_cvt_pk_bf16_f32 + single b32 LDS store
    const long xbase = ((long)(b * CIN_) * H_ + h) * W_;
    for (int idx = tid; idx < (CIN_ / 2) * XQ_; idx += 256) {
        int cp = idx / XQ_;                      // cin pair index
        int q  = idx - cp * XQ_;                 // x position index (gp = p0-1+q)
        int gp = p0 - 1 + q;
        int ci = cp * 2;
        float v0 = 0.0f, v1 = 0.0f;
        if (gp >= 0 && gp < W_) {
            v0 = x[xbase + (long)ci * (H_ * W_) + gp];
            v1 = x[xbase + (long)(ci + 1) * (H_ * W_) + gp];
        }
        v2bf p; p[0] = (__bf16)v0; p[1] = (__bf16)v1;
        *(v2bf*)&sxT[q * CSTR_ + ci] = p;
    }

    // ---- stage 0b: weights -> LDS, pre-swizzled into A-fragment layout -----
    // 16-bit 16x32 A layout: lane half shifts K by +8; VGPR v<4 -> K 2v,
    // v>=4 -> K 16+2(v-4); K-index = tap*64 + cin.
    for (int idx = tid; idx < COUT_ * CIN_ * 3; idx += 256) {
        int j  = idx & 15;
        int ln = (idx >> 4) & 31;
        int rest = idx >> 9;
        int kk = rest % 6;
        int ct = rest / 6;
        int v    = j >> 1;
        int lohi = j & 1;
        int klo  = (v < 4 ? 2 * v : 16 + 2 * (v - 4)) + 8 * (ln >> 4);
        int kidx = kk * 32 + klo + lohi;
        int tap  = kidx >> 6;
        int cin  = kidx & 63;
        int cout = ct * 16 + (ln & 15);
        sA[idx] = (__bf16)wgt[(cout * CIN_ + cin) * 3 + tap];
    }
    __syncthreads();

    // ---- stage 0c: hoist this wave's A fragments + bias into registers -----
    const int wave = tid >> 5;                   // 8 wave32
    const int lane = tid & 31;
    const int half = lane >> 4;
    const int n    = lane & 15;
    const int ct   = wave & 3;                   // one cout-tile per wave
    const int ptb  = (wave < 4) ? 0 : 5;         // 5 pos-tiles per wave

    v16bf afrag[6];
    #pragma unroll
    for (int kk = 0; kk < 6; ++kk)
        afrag[kk] = *(const v16bf*)&sA[((ct * 6 + kk) * 32 + lane) * 16];

    float bvals[8];
    #pragma unroll
    for (int r = 0; r < 8; ++r)
        bvals[r] = bias[ct * 16 + 8 * half + r];
    __syncthreads();                             // sA dead; sy may be written

    // ---- stage 1: WMMA conv -> sy -------------------------------------------
    #pragma unroll
    for (int pp = 0; pp < 5; ++pp) {
        const int pt  = ptb + pp;
        const int pos = pt * 16 + n;
        const __bf16* b0 = &sxT[pos * CSTR_ + 8 * half];

        // Batch all 12 B-fragment loads into distinct registers so the
        // in-order DS returns allow staged (partial) dscnt waits.
        v8bf bl[6], bh[6];
        #pragma unroll
        for (int kk = 0; kk < 6; ++kk) {
            bl[kk] = *(const v8bf*)(b0 + BOFF(kk));
            bh[kk] = *(const v8bf*)(b0 + BOFF(kk) + 16);
        }
        v8f acc = {};
        #pragma unroll
        for (int kk = 0; kk < 6; ++kk) {
            v16bf bv = __builtin_shufflevector(bl[kk], bh[kk],
                        0,1,2,3,4,5,6,7,8,9,10,11,12,13,14,15);
            acc = __builtin_amdgcn_wmma_f32_16x16x32_bf16(
                      false, afrag[kk], false, bv, (short)0, acc, false, false);
        }
        // D layout: lanes0-15 M=r, lanes16-31 M=r+8 -> 8 consecutive cout
        if (pos < YROWS_) {
            const int gp = p0 + pos;
            const bool valid = (gp >= 0 && gp < W_);
            v4f lovals, hivals;
            #pragma unroll
            for (int r = 0; r < 4; ++r) {
                lovals[r] = valid ? (acc[r]     + bvals[r])     : 0.0f;
                hivals[r] = valid ? (acc[r + 4] + bvals[r + 4]) : 0.0f;
            }
            float* dst = &sy[pos * YSTR_ + ct * 16 + 8 * half];
            *(v4f*)(dst)     = lovals;
            *(v4f*)(dst + 4) = hivals;
        }
    }
    __syncthreads();

    // ---- stage 2: folded polyphase up-FIR -> lrelu -> down-FIR -------------
    const float a0 = 4.0f * upf[4], a1 = 4.0f * upf[3], a2 = 4.0f * upf[2],
                a3 = 4.0f * upf[1], a4 = 4.0f * upf[0];
    const float d0 = dnf[3], d1 = dnf[2], d2 = dnf[1], d3 = dnf[0];

    const long obase = ((long)(b * COUT_) * H_ + h) * W_ + tile * TILE_;
    for (int idx = tid; idx < COUT_ * TILE_; idx += 256) {
        int c  = idx / TILE_;
        int il = idx - c * TILE_;
        int li = il + 8;                         // y row of output position i
        float ym1 = sy[(li - 1) * YSTR_ + c];
        float y0  = sy[(li    ) * YSTR_ + c];
        float y1  = sy[(li + 1) * YSTR_ + c];
        float y2  = sy[(li + 2) * YSTR_ + c];
        float u0 = a1 * ym1 + a3 * y0;
        float u1 = a0 * ym1 + a2 * y0 + a4 * y1;
        float u2 = a1 * y0  + a3 * y1;
        float u3 = a0 * y0  + a2 * y1 + a4 * y2;
        u0 = (u0 >= 0.0f) ? u0 : SLOPE_ * u0;
        u1 = (u1 >= 0.0f) ? u1 : SLOPE_ * u1;
        u2 = (u2 >= 0.0f) ? u2 : SLOPE_ * u2;
        u3 = (u3 >= 0.0f) ? u3 : SLOPE_ * u3;
        out[obase + (long)c * (H_ * W_) + il] = u0 * d0 + u1 * d1 + u2 * d2 + u3 * d3;
    }
}

extern "C" void kernel_launch(void* const* d_in, const int* in_sizes, int n_in,
                              void* d_out, int out_size, void* d_ws, size_t ws_size,
                              hipStream_t stream) {
    (void)in_sizes; (void)n_in; (void)out_size; (void)d_ws; (void)ws_size;
    const float* x    = (const float*)d_in[0];
    const float* wgt  = (const float*)d_in[1];
    const float* bias = (const float*)d_in[2];
    const float* upf  = (const float*)d_in[3];
    const float* dnf  = (const float*)d_in[4];
    float* out = (float*)d_out;

    dim3 grid(W_ / TILE_, H_, B_);   // 4 x 64 x 8 = 2048 workgroups
    fused_conv_upfirdn_kernel<<<grid, 256, 0, stream>>>(x, wgt, bias, upf, dnf, out);
}
